// RGLRU_61546881352303
// MI455X (gfx1250) — matrix operand order repaired and checked by reading, not compiled, not run on hardware
//
#include <hip/hip_runtime.h>

// ---------------------------------------------------------------------------
// RG-LRU (Griffin) block for MI455X / gfx1250, wave32 + WMMA bf16 +
// double-buffered global_load_async_to_lds_b128 tile staging (ASYNCcnt).
// B=4, S=4096, D_IN=D_H=D_OUT=1024, NB=8 blocks of 128.
// ---------------------------------------------------------------------------

#define DHID   1024
#define BATCH  4
#define SEQ    4096
#define DIN    1024
#define DOUTN  1024
#define NBLK   8
#define BD     128
#define NCH    32               // scan chunks
#define CLEN   (SEQ / NCH)      // 128 steps per chunk

typedef __attribute__((ext_vector_type(16))) __bf16 bf16x16;
typedef __attribute__((ext_vector_type(8)))  __bf16 bf16x8;
typedef __attribute__((ext_vector_type(8)))  float  f32x8;

static __device__ __forceinline__ unsigned short f32_to_bf16_bits(float f) {
  union { float f; unsigned int u; } cv; cv.f = f;
  unsigned int u = cv.u;
  u += 0x7FFFu + ((u >> 16) & 1u);          // round-to-nearest-even
  return (unsigned short)(u >> 16);
}

static __device__ __forceinline__ bf16x16 frag_cat(bf16x8 lo, bf16x8 hi) {
  return __builtin_shufflevector(lo, hi, 0,1,2,3,4,5,6,7,8,9,10,11,12,13,14,15);
}

// 32-bit LDS byte address from a generic shared pointer (LDS aperture keeps
// the workgroup-relative offset in addr[31:0]).
static __device__ __forceinline__ unsigned lds_addr32(const void* p) {
  return (unsigned)(unsigned long long)p;
}

// Two async b128 copies global->LDS covering 32 contiguous bytes.
// IOFFSET is added to BOTH the LDS and global address (ISA 10.7.4), so one
// LDS-address VGPR + one global-offset VGPR serve both halves.
static __device__ __forceinline__ void async_ld32B(unsigned lds, unsigned voff,
                                                   unsigned long long sbase) {
  asm volatile(
      "global_load_async_to_lds_b128 %0, %1, %2\n\t"
      "global_load_async_to_lds_b128 %0, %1, %2 offset:16"
      :: "v"(lds), "v"(voff), "s"(sbase) : "memory");
}

static __device__ __forceinline__ void wait_async_le4() {
  asm volatile("s_wait_asynccnt 0x4" ::: "memory");
}
static __device__ __forceinline__ void wait_async_le6() {
  asm volatile("s_wait_asynccnt 0x6" ::: "memory");
}
static __device__ __forceinline__ void wait_async_0() {
  asm volatile("s_wait_asynccnt 0x0" ::: "memory");
}

// ---------------------------------------------------------------------------
// K0a: fp32 -> bf16 elementwise convert (4 elems / thread)
// ---------------------------------------------------------------------------
__global__ void __launch_bounds__(256) k_cvt_bf16(const float* __restrict__ in,
                                                  unsigned short* __restrict__ out,
                                                  int n) {
  int i = (blockIdx.x * 256 + threadIdx.x) * 4;
  if (i + 3 < n) {
    float4 v = *reinterpret_cast<const float4*>(in + i);
    ushort4 o;
    o.x = f32_to_bf16_bits(v.x);
    o.y = f32_to_bf16_bits(v.y);
    o.z = f32_to_bf16_bits(v.z);
    o.w = f32_to_bf16_bits(v.w);
    *reinterpret_cast<ushort4*>(out + i) = o;
  }
}

// ---------------------------------------------------------------------------
// K0b: fp32 [R][C] -> bf16 [C][R] tiled transpose-convert (batched via z)
// ---------------------------------------------------------------------------
__global__ void __launch_bounds__(256) k_transpose_cvt(const float* __restrict__ in,
                                                       unsigned short* __restrict__ out,
                                                       int R, int C) {
  __shared__ float t[32][33];
  const size_t base = (size_t)blockIdx.z * R * C;
  const int r0 = blockIdx.y * 32, c0 = blockIdx.x * 32;
  const int tx = threadIdx.x & 31, ty = threadIdx.x >> 5;   // ty 0..7
#pragma unroll
  for (int i = 0; i < 32; i += 8)
    t[ty + i][tx] = in[base + (size_t)(r0 + ty + i) * C + c0 + tx];
  __syncthreads();
#pragma unroll
  for (int i = 0; i < 32; i += 8)
    out[base + (size_t)(c0 + ty + i) * R + r0 + tx] = f32_to_bf16_bits(t[tx][ty + i]);
}

// ---------------------------------------------------------------------------
// WMMA GEMM: C[M][N] = A[M][K](bf16,row) * Bt[N][K](bf16,N-major) + bias[N]
// Block: 256 thr (8 waves), tile 128x128, K-step 32. Wave = 64x32 (4x2 WMMA).
// LDS tiles padded to stride 40 halves (80B) -> conflict-free b128 frag reads.
// Double-buffered async global->LDS staging (4 async ops/group/thread).
// ---------------------------------------------------------------------------
constexpr int BM = 128, BN = 128, BK = 32;
constexpr int LDT = BK + 8;   // 40

template <bool WRITE_BF16>
__global__ void __launch_bounds__(256) k_gemm_bf16(
    const unsigned short* __restrict__ A,   // [M][K]
    const unsigned short* __restrict__ Bt,  // [N][K]
    const float* __restrict__ bias,         // [N]
    float* __restrict__ C,                  // [M][N]
    unsigned short* __restrict__ Cbf,       // optional bf16 copy of C
    int M, int N, int K) {
  __shared__ __align__(16) unsigned short As[2][BM * LDT];
  __shared__ __align__(16) unsigned short Bs[2][BN * LDT];

  const int tid  = threadIdx.x;
  const int lane = tid & 31, wave = tid >> 5;
  const int wm = wave >> 2;        // 0..1  (64 rows)
  const int wn = wave & 3;         // 0..3  (32 cols)
  const int half = lane >> 4, l16 = lane & 15;
  const int m0 = blockIdx.x * BM;
  const int n0 = blockIdx.y * BN;

  f32x8 acc[4][2] = {};

  const int ldr = tid >> 1;            // 0..127
  const int ldc = (tid & 1) * 16;      // 0 / 16

  const unsigned ldsA0 = lds_addr32(&As[0][ldr * LDT + ldc]);
  const unsigned ldsA1 = lds_addr32(&As[1][ldr * LDT + ldc]);
  const unsigned ldsB0 = lds_addr32(&Bs[0][ldr * LDT + ldc]);
  const unsigned ldsB1 = lds_addr32(&Bs[1][ldr * LDT + ldc]);
  const unsigned voffA = (unsigned)(((unsigned)(m0 + ldr) * (unsigned)K + ldc) * 2u);
  const unsigned voffB = (unsigned)(((unsigned)(n0 + ldr) * (unsigned)K + ldc) * 2u);
  const unsigned long long sA = (unsigned long long)A;
  const unsigned long long sB = (unsigned long long)Bt;

  // prologue: stage K-step 0 into buffer 0
  async_ld32B(ldsA0, voffA, sA);
  async_ld32B(ldsB0, voffB, sB);

  int cur = 0;
  for (int k0 = 0; k0 < K; k0 += BK) {
    if (k0 + BK < K) {   // prefetch next K-step into the other buffer
      const unsigned kb = (unsigned)(k0 + BK) * 2u;
      async_ld32B(cur ? ldsA0 : ldsA1, voffA + kb, sA);
      async_ld32B(cur ? ldsB0 : ldsB1, voffB + kb, sB);
      wait_async_le4();                // current group complete, next in flight
    } else {
      wait_async_0();
    }
    __syncthreads();

    // A fragment: lane row = m, elems = K[16h .. 16h+15] contiguous
    bf16x16 af[4];
#pragma unroll
    for (int mt = 0; mt < 4; ++mt) {
      const int row = wm * 64 + mt * 16 + l16;
      bf16x8 lo = *reinterpret_cast<const bf16x8*>(&As[cur][row * LDT + 16 * half]);
      bf16x8 hi = *reinterpret_cast<const bf16x8*>(&As[cur][row * LDT + 16 * half + 8]);
      af[mt] = frag_cat(lo, hi);
    }
    // B fragment: lane col = n, elems = K[16h .. 16h+15] contiguous
    bf16x16 bfr[2];
#pragma unroll
    for (int nt = 0; nt < 2; ++nt) {
      const int col = wn * 32 + nt * 16 + l16;
      bf16x8 lo = *reinterpret_cast<const bf16x8*>(&Bs[cur][col * LDT + 16 * half]);
      bf16x8 hi = *reinterpret_cast<const bf16x8*>(&Bs[cur][col * LDT + 16 * half + 8]);
      bfr[nt] = frag_cat(lo, hi);
    }
#pragma unroll
    for (int mt = 0; mt < 4; ++mt)
#pragma unroll
      for (int nt = 0; nt < 2; ++nt)
        acc[mt][nt] = __builtin_amdgcn_wmma_f32_16x16x32_bf16(
            false, af[mt], false, bfr[nt], (short)0, acc[mt][nt], false, false);
    __syncthreads();   // all waves done reading `cur` before it is re-staged
    cur ^= 1;
  }

  // epilogue: C tile layout -> lane(n=l16, rows r+8*half)
#pragma unroll
  for (int mt = 0; mt < 4; ++mt)
#pragma unroll
    for (int nt = 0; nt < 2; ++nt) {
      const int col = n0 + wn * 32 + nt * 16 + l16;
      const float bv = bias[col];
      const int rowb = m0 + wm * 64 + mt * 16 + 8 * half;
#pragma unroll
      for (int r = 0; r < 8; ++r) {
        float v = acc[mt][nt][r] + bv;
        size_t idx = (size_t)(rowb + r) * N + col;
        C[idx] = v;
        if constexpr (WRITE_BF16) Cbf[idx] = f32_to_bf16_bits(v);
      }
    }
}

// ---------------------------------------------------------------------------
// K2: block-diagonal gate GEMMs (both gates in one K-loop) fused with the
// full RG-LRU elementwise epilogue: a = exp(-8*sigmoid(ga)*softplus(c)),
// gx = sigmoid(gxp) * xproj * sqrt(1 - a^2)
// grid.x = M/128 token tiles, grid.y = block index (0..7)
// ---------------------------------------------------------------------------
__global__ void __launch_bounds__(256) k_gates(
    const unsigned short* __restrict__ Xbf,   // [BS][DH] bf16 (xproj)
    const float* __restrict__ Xf32,           // [BS][DH] fp32 (xproj)
    const unsigned short* __restrict__ WgxT,  // [NB][BD][BD]  (o-major)
    const unsigned short* __restrict__ WgaT,  // [NB][BD][BD]
    const float* __restrict__ bgx, const float* __restrict__ bga,
    const float* __restrict__ arp,            // [DH] a_real_param
    float* __restrict__ Aout, float* __restrict__ GXout) {
  __shared__ __align__(16) unsigned short As[2][BM * LDT];
  __shared__ __align__(16) unsigned short BXs[2][BD * LDT];
  __shared__ __align__(16) unsigned short BAs[2][BD * LDT];

  const int tid = threadIdx.x, lane = tid & 31, wave = tid >> 5;
  const int wm = wave >> 2, wn = wave & 3;
  const int half = lane >> 4, l16 = lane & 15;
  const int m0  = blockIdx.x * BM;
  const int blk = blockIdx.y;

  f32x8 accX[4][2] = {};
  f32x8 accA[4][2] = {};

  const int ldr = tid >> 1, ldc = (tid & 1) * 16;
  const unsigned ldsA0 = lds_addr32(&As[0][ldr * LDT + ldc]);
  const unsigned ldsA1 = lds_addr32(&As[1][ldr * LDT + ldc]);
  const unsigned ldsX0 = lds_addr32(&BXs[0][ldr * LDT + ldc]);
  const unsigned ldsX1 = lds_addr32(&BXs[1][ldr * LDT + ldc]);
  const unsigned ldsG0 = lds_addr32(&BAs[0][ldr * LDT + ldc]);
  const unsigned ldsG1 = lds_addr32(&BAs[1][ldr * LDT + ldc]);
  const unsigned voffA = (unsigned)(((unsigned)(m0 + ldr) * DHID + blk * BD + ldc) * 2u);
  const unsigned voffW = (unsigned)(((unsigned)(blk * BD + ldr) * BD + ldc) * 2u);
  const unsigned long long sX = (unsigned long long)Xbf;
  const unsigned long long sWx = (unsigned long long)WgxT;
  const unsigned long long sWa = (unsigned long long)WgaT;

  async_ld32B(ldsA0, voffA, sX);
  async_ld32B(ldsX0, voffW, sWx);
  async_ld32B(ldsG0, voffW, sWa);

  int cur = 0;
  for (int k0 = 0; k0 < BD; k0 += BK) {
    if (k0 + BK < BD) {
      const unsigned kb = (unsigned)(k0 + BK) * 2u;
      async_ld32B(cur ? ldsA0 : ldsA1, voffA + kb, sX);
      async_ld32B(cur ? ldsX0 : ldsX1, voffW + kb, sWx);
      async_ld32B(cur ? ldsG0 : ldsG1, voffW + kb, sWa);
      wait_async_le6();
    } else {
      wait_async_0();
    }
    __syncthreads();

    bf16x16 af[4];
#pragma unroll
    for (int mt = 0; mt < 4; ++mt) {
      const int row = wm * 64 + mt * 16 + l16;
      bf16x8 lo = *reinterpret_cast<const bf16x8*>(&As[cur][row * LDT + 16 * half]);
      bf16x8 hi = *reinterpret_cast<const bf16x8*>(&As[cur][row * LDT + 16 * half + 8]);
      af[mt] = frag_cat(lo, hi);
    }
    bf16x16 bx[2], ba[2];
#pragma unroll
    for (int nt = 0; nt < 2; ++nt) {
      const int col = wn * 32 + nt * 16 + l16;
      bf16x8 lo = *reinterpret_cast<const bf16x8*>(&BXs[cur][col * LDT + 16 * half]);
      bf16x8 hi = *reinterpret_cast<const bf16x8*>(&BXs[cur][col * LDT + 16 * half + 8]);
      bx[nt] = frag_cat(lo, hi);
      lo = *reinterpret_cast<const bf16x8*>(&BAs[cur][col * LDT + 16 * half]);
      hi = *reinterpret_cast<const bf16x8*>(&BAs[cur][col * LDT + 16 * half + 8]);
      ba[nt] = frag_cat(lo, hi);
    }
#pragma unroll
    for (int mt = 0; mt < 4; ++mt)
#pragma unroll
      for (int nt = 0; nt < 2; ++nt) {
        accX[mt][nt] = __builtin_amdgcn_wmma_f32_16x16x32_bf16(
            false, af[mt], false, bx[nt], (short)0, accX[mt][nt], false, false);
        accA[mt][nt] = __builtin_amdgcn_wmma_f32_16x16x32_bf16(
            false, af[mt], false, ba[nt], (short)0, accA[mt][nt], false, false);
      }
    __syncthreads();
    cur ^= 1;
  }

#pragma unroll
  for (int mt = 0; mt < 4; ++mt)
#pragma unroll
    for (int nt = 0; nt < 2; ++nt) {
      const int col = blk * BD + wn * 32 + nt * 16 + l16;
      const float bxv = bgx[col], bav = bga[col];
      const float p = arp[col];
      const float c = fmaxf(p, 0.f) + log1pf(expf(-fabsf(p)));  // softplus
      const int rowb = m0 + wm * 64 + mt * 16 + 8 * half;
#pragma unroll
      for (int r = 0; r < 8; ++r) {
        const float gx_ = 1.f / (1.f + expf(-(accX[mt][nt][r] + bxv)));
        const float ga_ = 1.f / (1.f + expf(-(accA[mt][nt][r] + bav)));
        const float log_a = -8.f * ga_ * c;
        const float av = expf(log_a);
        const float om = 1.f - expf(2.f * log_a);
        const float mult = gx_ * sqrtf(fmaxf(om, 0.f));
        const size_t idx = (size_t)(rowb + r) * DHID + col;
        Aout[idx]  = av;
        GXout[idx] = mult * Xf32[idx];
      }
    }
}

// ---------------------------------------------------------------------------
// Chunked linear scan: h_t = a_t*h_{t-1} + gx_t
// thread = (b, chunk, channel); 131072 threads in phase 1/3, 4096 in phase 2
// ---------------------------------------------------------------------------
__global__ void __launch_bounds__(256) k_scan_agg(
    const float* __restrict__ A, const float* __restrict__ GX,
    float* __restrict__ aProd, float* __restrict__ xAgg) {
  const int g = blockIdx.x * 256 + threadIdx.x;
  const int c = g & (DHID - 1);
  const int rest = g >> 10;
  const int chunk = rest & (NCH - 1);
  const int b = rest >> 5;
  size_t base = ((size_t)b * SEQ + (size_t)chunk * CLEN) * DHID + c;
  float ap = 1.f, xa = 0.f;
  for (int t = 0; t < CLEN; ++t) {
    const float at = A[base];
    const float gt = GX[base];
    ap = at * ap;
    xa = at * xa + gt;
    base += DHID;
  }
  const size_t ci = ((size_t)b * NCH + chunk) * DHID + c;
  aProd[ci] = ap;
  xAgg[ci]  = xa;
}

__global__ void __launch_bounds__(256) k_scan_carry(
    const float* __restrict__ aProd, const float* __restrict__ xAgg,
    const float* __restrict__ h0, float* __restrict__ hIn) {
  const int g = blockIdx.x * 256 + threadIdx.x;   // < B*DH
  const int c = g & (DHID - 1);
  const int b = g >> 10;
  float h = h0[(size_t)b * DHID + c];
  for (int ch = 0; ch < NCH; ++ch) {
    const size_t ci = ((size_t)b * NCH + ch) * DHID + c;
    hIn[ci] = h;                     // carry-in (h before this chunk)
    h = aProd[ci] * h + xAgg[ci];
  }
}

__global__ void __launch_bounds__(256) k_scan_final(
    const float* __restrict__ A, const float* __restrict__ GX,
    const float* __restrict__ hIn,
    unsigned short* __restrict__ Hbf, float* __restrict__ hLast) {
  const int g = blockIdx.x * 256 + threadIdx.x;
  const int c = g & (DHID - 1);
  const int rest = g >> 10;
  const int chunk = rest & (NCH - 1);
  const int b = rest >> 5;
  const size_t ci = ((size_t)b * NCH + chunk) * DHID + c;
  size_t base = ((size_t)b * SEQ + (size_t)chunk * CLEN) * DHID + c;
  float h = hIn[ci];
  for (int t = 0; t < CLEN; ++t) {
    h = A[base] * h + GX[base];
    Hbf[base] = f32_to_bf16_bits(h);
    base += DHID;
  }
  if (chunk == NCH - 1) hLast[(size_t)b * DHID + c] = h;
}

// ---------------------------------------------------------------------------
// Host-side orchestration (graph-capture safe: launches only)
// ---------------------------------------------------------------------------
extern "C" void kernel_launch(void* const* d_in, const int* in_sizes, int n_in,
                              void* d_out, int out_size, void* d_ws, size_t ws_size,
                              hipStream_t stream) {
  (void)in_sizes; (void)n_in; (void)out_size; (void)ws_size;
  const float* x    = (const float*)d_in[0];
  const float* h0   = (const float*)d_in[1];
  const float* Win  = (const float*)d_in[2];
  const float* bin  = (const float*)d_in[3];
  const float* Wgx  = (const float*)d_in[4];
  const float* bgx  = (const float*)d_in[5];
  const float* Wga  = (const float*)d_in[6];
  const float* bga  = (const float*)d_in[7];
  const float* arp  = (const float*)d_in[8];
  const float* Wout = (const float*)d_in[9];
  const float* bout = (const float*)d_in[10];
  float* out = (float*)d_out;

  char* ws = (char*)d_ws;
  const size_t BS = (size_t)BATCH * SEQ;   // 16384 token rows
  size_t off = 0;
  auto alloc = [&](size_t bytes) {
    size_t o = off; off += (bytes + 255) & ~(size_t)255; return o;
  };
  unsigned short* xbf     = (unsigned short*)(ws + alloc(BS * DIN * 2));
  unsigned short* WinT    = (unsigned short*)(ws + alloc((size_t)DIN * DHID * 2));
  unsigned short* WgxT    = (unsigned short*)(ws + alloc((size_t)NBLK * BD * BD * 2));
  unsigned short* WgaT    = (unsigned short*)(ws + alloc((size_t)NBLK * BD * BD * 2));
  unsigned short* WoutT   = (unsigned short*)(ws + alloc((size_t)DHID * DOUTN * 2));
  float*          xproj   = (float*)(ws + alloc(BS * DHID * 4));
  unsigned short* xprojbf = (unsigned short*)(ws + alloc(BS * DHID * 2));
  float*          aBuf    = (float*)(ws + alloc(BS * DHID * 4));
  float*          gxBuf   = (float*)(ws + alloc(BS * DHID * 4));
  float*          aProd   = (float*)(ws + alloc((size_t)BATCH * NCH * DHID * 4));
  float*          xAgg    = (float*)(ws + alloc((size_t)BATCH * NCH * DHID * 4));
  float*          hIn     = (float*)(ws + alloc((size_t)BATCH * NCH * DHID * 4));
  unsigned short* hbf     = xbf;   // x_bf16 is dead after GEMM1 -> reuse for h

  // K0: precision conversion / weight transposition (one-time, ~100MB total)
  const int nx = (int)(BS * DIN);
  k_cvt_bf16<<<nx / 1024, 256, 0, stream>>>(x, xbf, nx);
  k_transpose_cvt<<<dim3(DHID / 32, DIN / 32, 1),  256, 0, stream>>>(Win,  WinT,  DIN,  DHID);
  k_transpose_cvt<<<dim3(BD / 32,   BD / 32, NBLK), 256, 0, stream>>>(Wgx,  WgxT,  BD,   BD);
  k_transpose_cvt<<<dim3(BD / 32,   BD / 32, NBLK), 256, 0, stream>>>(Wga,  WgaT,  BD,   BD);
  k_transpose_cvt<<<dim3(DOUTN / 32, DHID / 32, 1), 256, 0, stream>>>(Wout, WoutT, DHID, DOUTN);

  // K1: xproj = x @ W_in + b_in  (fp32 + bf16 outputs)
  k_gemm_bf16<true><<<dim3((unsigned)(BS / BM), DHID / BN), 256, 0, stream>>>(
      xbf, WinT, bin, xproj, xprojbf, (int)BS, DHID, DIN);

  // K2: block-diag gate GEMMs + fused RG-LRU elementwise -> a, gx
  k_gates<<<dim3((unsigned)(BS / BM), NBLK), 256, 0, stream>>>(
      xprojbf, xproj, WgxT, WgaT, bgx, bga, arp, aBuf, gxBuf);

  // K3-K5: chunked scan (h0 folded in as chunk-0 carry), bf16 h + fp32 h_last
  k_scan_agg<<<(BATCH * NCH * DHID) / 256, 256, 0, stream>>>(aBuf, gxBuf, aProd, xAgg);
  k_scan_carry<<<(BATCH * DHID) / 256, 256, 0, stream>>>(aProd, xAgg, h0, hIn);
  k_scan_final<<<(BATCH * NCH * DHID) / 256, 256, 0, stream>>>(
      aBuf, gxBuf, hIn, hbf, out + BS * (size_t)DOUTN);

  // K6: out = h @ W_out + b_out
  k_gemm_bf16<false><<<dim3((unsigned)(BS / BM), DOUTN / BN), 256, 0, stream>>>(
      hbf, WoutT, bout, out, nullptr, (int)BS, DOUTN, DHID);
}
